// WaveletMambaBranch_21053929685294
// MI455X (gfx1250) — compile-verified, matrix operand order checked by reading, not run on hardware
//
#include <hip/hip_runtime.h>
#include <hip/hip_bf16.h>
#include <stdint.h>

// ---------------------------------------------------------------------------
// WaveletMambaBranch on gfx1250 (MI455X), wave32.
// Pipeline: DWT -> LL layernorm -> in_proj (WMMA bf16) -> dwconv+SiLU ->
//   gather 4 scan dirs -> x_proj (WMMA bf16) -> 3-phase chunked selective scan
//   (x_dbl chunk staged into LDS via TDM tensor_load_to_lds) -> merge dirs ->
//   LN*gate -> out_proj (WMMA bf16) ; SE on high bands ; IDWT.
// ---------------------------------------------------------------------------

typedef __attribute__((ext_vector_type(16))) __bf16 v16bf;
typedef __attribute__((ext_vector_type(8)))  float  v8f;
typedef __attribute__((ext_vector_type(4)))  unsigned int v4u;
typedef __attribute__((ext_vector_type(8)))  int  v8i_t;
typedef __attribute__((ext_vector_type(4)))  int  v4i_t;

#define B_SZ    4
#define CDIM    96
#define D_INNER 192
#define NSTATE  16
#define DTRANK  6
#define KDIR    4
#define H2      64
#define W2      64
#define LSEQ    4096
#define BL      16384     // B_SZ * LSEQ
#define NPROJ   48        // 38 (6 dt + 16 B + 16 C) padded to 48
#define NCHUNK  64
#define CHUNK   64

// ---------------- helpers ----------------
__device__ __forceinline__ unsigned short f2bf(float f) {
  unsigned int u = __float_as_uint(f);
  u += 0x7FFFu + ((u >> 16) & 1u);          // round-to-nearest-even
  return (unsigned short)(u >> 16);
}
__device__ __forceinline__ float bf2f(unsigned short v) {
  return __uint_as_float(((unsigned int)v) << 16);
}
__device__ __forceinline__ float softplusf(float x) {
  return (x > 20.f) ? x : log1pf(__expf(x));
}
__device__ __forceinline__ float siluf(float x) {
  return x / (1.f + __expf(-x));
}
__device__ __forceinline__ int reflect128(int p) {
  p = (p < 0) ? -p : p;
  return (p > 127) ? (254 - p) : p;
}

// --------- TDM: 1-D contiguous f32 tile, global -> LDS (CDNA5 ISA §8) -------
// D# group0: count=1 | lds_addr | global_addr[56:0] | type=2 ("image")
// D# group1: data_size=2 (4B), tensor_dim0=n, tensor_dim1=1, tile_dim0=n,
//            tensor_dim0_stride=n. Groups 2/3 zero (<=2D tensor).
__device__ __forceinline__ void tdm_load_f32_to_lds(void* lds_dst,
                                                    const void* gsrc,
                                                    unsigned int nelem) {
  unsigned int lds_off = (unsigned int)(size_t)lds_dst;   // low 32b = LDS addr
  unsigned long long ga = (unsigned long long)(size_t)gsrc;
  v4u g0;
  g0[0] = 1u;                                             // count=1, user mode
  g0[1] = lds_off;
  g0[2] = (unsigned int)(ga & 0xFFFFFFFFu);
  g0[3] = (unsigned int)((ga >> 32) & 0x01FFFFFFu) | (2u << 30);
  v8i_t g1;
  g1[0] = (int)(2u << 16);                                // data_size = 4B
  g1[1] = (int)((nelem & 0xFFFFu) << 16);                 // tensor_dim0 lo16
  g1[2] = (int)(((nelem >> 16) & 0xFFFFu) | (1u << 16));  // dim0 hi | dim1=1
  g1[3] = (int)((nelem & 0xFFFFu) << 16);                 // tile_dim0
  g1[4] = 0;                                              // tile_dim1/2 unused
  g1[5] = (int)nelem;                                     // dim0_stride lo32
  g1[6] = 0;
  g1[7] = 0;
  v4i_t z4 = {0, 0, 0, 0};
#if defined(__clang_major__) && (__clang_major__ >= 23)
  v8i_t z8 = {0, 0, 0, 0, 0, 0, 0, 0};
  __builtin_amdgcn_tensor_load_to_lds(g0, g1, z4, z4, z8, 0);
#else
  __builtin_amdgcn_tensor_load_to_lds(g0, g1, z4, z4, 0);
#endif
}

// ---------------- 0: weight prep -> WMMA-fragment-native packed bf16 --------
// Packed B layout: pk[((n*KT + kt)*2 + half)*16 + e] = W^T[kmap][n]
//   kmap = kt*32 + half*8 + (e<8 ? e : e+8)     (ISA 7.12.2 B-fragment order)
__global__ void prep_weights(const float* __restrict__ in_proj_w,
                             const float* __restrict__ x_proj_w,
                             const float* __restrict__ out_proj_w,
                             unsigned short* __restrict__ wip,   // N=384,KT=3
                             unsigned short* __restrict__ wxp,   // 4 x N=48,KT=6
                             unsigned short* __restrict__ wop) { // N=96,KT=6
  int i = blockIdx.x * 256 + threadIdx.x;
  if (i < 36864) {                       // in_proj: K=96, N=384
    int e = i & 15, half = (i >> 4) & 1;
    int kt = (i >> 5) % 3;
    int n  = i / 96;
    int km = kt * 32 + half * 8 + ((e < 8) ? e : e + 8);
    wip[i] = f2bf(in_proj_w[n * 96 + km]);
  } else if (i < 73728) {                // x_proj: 4 dirs, K=192, N=48(pad 38)
    int j = i - 36864;
    int e = j & 15, half = (j >> 4) & 1;
    int kt = (j >> 5) % 6;
    int rest = j / 192;
    int n = rest % 48;
    int k = rest / 48;
    int km = kt * 32 + half * 8 + ((e < 8) ? e : e + 8);
    float v = (n < 38) ? x_proj_w[(size_t)(k * 38 + n) * 192 + km] : 0.f;
    wxp[j] = f2bf(v);
  } else if (i < 92160) {                // out_proj: K=192, N=96
    int j = i - 73728;
    int e = j & 15, half = (j >> 4) & 1;
    int kt = (j >> 5) % 6;
    int n  = j / 192;
    int km = kt * 32 + half * 8 + ((e < 8) ? e : e + 8);
    wop[j] = f2bf(out_proj_w[n * 192 + km]);
  }
}

// ---------------- 1: DWT (6x6, stride 2, reflect pad 2) ----------------
__global__ void dwt_kernel(const float* __restrict__ x,
                           const float* __restrict__ dec_f,
                           float* __restrict__ x_dwt) {
  int idx = blockIdx.x * 256 + threadIdx.x;
  if (idx >= B_SZ * CDIM * 4 * LSEQ) return;
  int ox = idx & 63;
  int oy = (idx >> 6) & 63;
  int s  = (idx >> 12) & 3;
  int c  = (idx >> 14) % CDIM;
  int b  = idx / (CDIM << 14);
  const float* f  = dec_f + (size_t)(c * 4 + s) * 36;
  const float* xb = x + (size_t)(b * CDIM + c) * 16384;
  float acc = 0.f;
  #pragma unroll
  for (int i = 0; i < 6; i++) {
    int p = reflect128(2 * oy + i - 2);
    const float* row = xb + (size_t)p * 128;
    #pragma unroll
    for (int j = 0; j < 6; j++) {
      int q = reflect128(2 * ox + j - 2);
      acc += f[i * 6 + j] * row[q];
    }
  }
  x_dwt[idx] = acc;
}

// ---------------- 2: layernorm of LL band -> bf16 (BL,96) ----------------
__global__ void ll_layernorm(const float* __restrict__ x_dwt,
                             const float* __restrict__ ln_w,
                             const float* __restrict__ ln_b,
                             unsigned short* __restrict__ A_ll) {
  int tid = blockIdx.x * 256 + threadIdx.x;
  if (tid >= BL) return;
  int b = tid >> 12;
  int l = tid & 4095;
  const float* base = x_dwt + (size_t)b * CDIM * 4 * LSEQ + l;
  float s = 0.f, s2 = 0.f;
  for (int c = 0; c < CDIM; c++) {
    float v = base[(size_t)c * (4 * LSEQ)];
    s += v; s2 += v * v;
  }
  float m   = s * (1.f / 96.f);
  float var = s2 * (1.f / 96.f) - m * m;
  float inv = rsqrtf(var + 1e-5f);
  for (int c = 0; c < CDIM; c++) {
    float v = base[(size_t)c * (4 * LSEQ)];
    A_ll[(size_t)tid * 96 + c] = f2bf((v - m) * inv * ln_w[c] + ln_b[c]);
  }
}

// ---------------- 3: bf16 WMMA GEMM, 64x16 tile per wave ----------
// A: (M,K) row-major bf16; Bp: packed fragment layout; C: (M,N) f32.
__global__ __launch_bounds__(128)
void gemm_bf16_wmma(const unsigned short* __restrict__ A,
                    const unsigned short* __restrict__ Bp,
                    float* __restrict__ C,
                    int MtilesSuper, int Ntiles, int Kdim,
                    int lda, int ldc,
                    int batch, long long sA, long long sB, long long sC) {
  int wid  = (blockIdx.x * blockDim.x + threadIdx.x) >> 5;
  int lane = threadIdx.x & 31;
  int tilesPB = MtilesSuper * Ntiles;
  int bt = wid / tilesPB;
  if (bt >= batch) return;
  int t  = wid % tilesPB;
  int m0 = (t / Ntiles) << 6;    // 64 rows per wave
  int n0 = (t % Ntiles) << 4;
  int KT = Kdim >> 5;
  const unsigned short* Ab = A  + (size_t)bt * sA;
  const unsigned short* Bb = Bp + (size_t)bt * sB;
  float* Cb = C + (size_t)bt * sC;

  int rlane = lane & 15;
  int half  = lane >> 4;
  int kb2   = half << 3;         // K sub-offset per lane-half (ISA 7.12.2)
  int col   = n0 + rlane;

  v8f acc[4];
  #pragma unroll
  for (int s = 0; s < 4; s++) {
    #pragma unroll
    for (int i = 0; i < 8; i++) acc[s][i] = 0.f;
  }

  union Frag { unsigned int u32[8]; unsigned short u16[16]; v16bf v; };

  for (int kk = 0; kk < Kdim; kk += 32) {
    int kt = kk >> 5;
    // B fragment: one contiguous 32B read per lane (packed at prep time)
    Frag bfr;
    const unsigned int* pb = (const unsigned int*)
        (Bb + (((size_t)col * KT + kt) * 2 + half) * 16);
    #pragma unroll
    for (int i = 0; i < 8; i++) bfr.u32[i] = pb[i];
    __builtin_prefetch(Ab + (size_t)(m0 + rlane) * lda + kk + 32, 0, 1);
    #pragma unroll
    for (int s = 0; s < 4; s++) {
      int row = m0 + s * 16 + rlane;
      Frag af;
      const unsigned int* pa0 =
          (const unsigned int*)(Ab + (size_t)row * lda + kk + kb2);
      const unsigned int* pa1 =
          (const unsigned int*)(Ab + (size_t)row * lda + kk + kb2 + 16);
      #pragma unroll
      for (int i = 0; i < 4; i++) { af.u32[i] = pa0[i]; af.u32[4 + i] = pa1[i]; }
      acc[s] = __builtin_amdgcn_wmma_f32_16x16x32_bf16(
          false, af.v, false, bfr.v, (short)0, acc[s], false, false);
    }
  }
  #pragma unroll
  for (int s = 0; s < 4; s++) {
    int rbase = m0 + s * 16 + (half << 3);
    #pragma unroll
    for (int r = 0; r < 8; r++)
      Cb[(size_t)(rbase + r) * ldc + col] = acc[s][r];
  }
}

// ---------------- 4: depthwise 3x3 conv + bias + SiLU ----------------
__global__ __launch_bounds__(192)
void conv_silu(const float* __restrict__ xz,      // (BL,384), xc = cols 0..191
               const float* __restrict__ conv_w,  // (192,1,3,3)
               const float* __restrict__ conv_b,
               float* __restrict__ xc_act) {      // (BL,192)
  int d  = threadIdx.x;
  int bl = blockIdx.x;
  int b  = bl >> 12;
  int l  = bl & 4095;
  int y0 = l >> 6, x0 = l & 63;
  float acc = conv_b[d];
  #pragma unroll
  for (int i = 0; i < 3; i++) {
    int yy = y0 + i - 1;
    if (yy < 0 || yy > 63) continue;
    #pragma unroll
    for (int j = 0; j < 3; j++) {
      int xx = x0 + j - 1;
      if (xx < 0 || xx > 63) continue;
      acc += conv_w[d * 9 + i * 3 + j] *
             xz[((size_t)(b << 12) + (yy << 6) + xx) * 384 + d];
    }
  }
  xc_act[(size_t)bl * 192 + d] = siluf(acc);
}

// ---------------- 5: gather 4 scan orders -> xs bf16 (K,B,L,D) ----------------
__global__ void gather_xs(const float* __restrict__ xc_act,
                          unsigned short* __restrict__ xs) {
  int idx = blockIdx.x * 256 + threadIdx.x;
  if (idx >= KDIR * BL * D_INNER) return;
  int d  = idx % 192;
  int s  = (idx / 192) & 4095;
  int kb = idx / (192 * 4096);
  int b  = kb & 3;
  int k  = kb >> 2;
  int t  = (k >= 2) ? (4095 - s) : s;
  int p  = (k & 1) ? (((t & 63) << 6) + (t >> 6)) : t;
  xs[idx] = f2bf(xc_act[((size_t)(b << 12) + p) * 192 + d]);
}

// ---------------- 6: scan phase 1: per-chunk A-product and local h_end -------
__global__ __launch_bounds__(192)
void scan_phase1(const float* __restrict__ x_dbl,        // (K,B,L,48)
                 const unsigned short* __restrict__ xs,  // (K,B,L,192)
                 const float* __restrict__ dt_w,         // (K,192,6)
                 const float* __restrict__ dt_b,         // (K,192)
                 const float* __restrict__ A_logs,       // (K*192,16)
                 float* __restrict__ chAH) {             // (ch,chunk,32)
  int bid = blockIdx.x;
  int chunk = bid & 63;
  int b = (bid >> 6) & 3;
  int k = bid >> 8;
  int d = threadIdx.x;
  __shared__ float sd[CHUNK * NPROJ];
  size_t dblbase = ((size_t)(k * 4 + b) * LSEQ + (size_t)chunk * CHUNK) * NPROJ;
  if (threadIdx.x < 32) {                         // wave 0 drives the TDM
    tdm_load_f32_to_lds(sd, x_dbl + dblbase, CHUNK * NPROJ);
    __builtin_amdgcn_s_wait_tensorcnt((short)0);
  }

  int kd = k * 192 + d;
  float dtw[6];
  #pragma unroll
  for (int r = 0; r < 6; r++) dtw[r] = dt_w[kd * 6 + r];
  float dtb = dt_b[kd];
  float Ad[16];
  #pragma unroll
  for (int n = 0; n < 16; n++) Ad[n] = -__expf(A_logs[(size_t)kd * 16 + n]);

  float h[16], ap[16];
  #pragma unroll
  for (int n = 0; n < 16; n++) { h[n] = 0.f; ap[n] = 1.f; }

  __syncthreads();                                // LDS tile visible to all

  size_t ubase = ((size_t)(k * 4 + b) * LSEQ + (size_t)chunk * CHUNK) * 192 + d;
  for (int t = 0; t < CHUNK; t++) {
    const float* row = sd + t * NPROJ;
    float xdt = dtb;
    #pragma unroll
    for (int r = 0; r < 6; r++) xdt += row[r] * dtw[r];
    float delta = softplusf(xdt);
    float u  = bf2f(xs[ubase + (size_t)t * 192]);
    float du = delta * u;
    #pragma unroll
    for (int n = 0; n < 16; n++) {
      float dA = __expf(delta * Ad[n]);
      h[n]  = dA * h[n] + du * row[6 + n];
      ap[n] *= dA;
    }
  }
  int ch = (k * 4 + b) * 192 + d;
  size_t cb = ((size_t)ch * NCHUNK + chunk) * 32;
  #pragma unroll
  for (int n = 0; n < 16; n++) { chAH[cb + n] = ap[n]; chAH[cb + 16 + n] = h[n]; }
}

// ---------------- 7: scan phase 2: sequential combine over chunks ------------
__global__ void scan_phase2(const float* __restrict__ chAH,
                            float* __restrict__ hin) {
  int ch = blockIdx.x * 256 + threadIdx.x;
  if (ch >= KDIR * B_SZ * D_INNER) return;
  float h[16];
  #pragma unroll
  for (int n = 0; n < 16; n++) h[n] = 0.f;
  for (int chunk = 0; chunk < NCHUNK; chunk++) {
    size_t hb = ((size_t)ch * NCHUNK + chunk) * 16;
    size_t cb = ((size_t)ch * NCHUNK + chunk) * 32;
    #pragma unroll
    for (int n = 0; n < 16; n++) {
      hin[hb + n] = h[n];
      h[n] = chAH[cb + n] * h[n] + chAH[cb + 16 + n];
    }
  }
}

// ---------------- 8: scan phase 3: full scan with correct h_in, emit y -------
__global__ __launch_bounds__(192)
void scan_phase3(const float* __restrict__ x_dbl,
                 const unsigned short* __restrict__ xs,
                 const float* __restrict__ dt_w,
                 const float* __restrict__ dt_b,
                 const float* __restrict__ A_logs,
                 const float* __restrict__ Ds,
                 const float* __restrict__ hin,
                 float* __restrict__ out_y) {            // (K,B,L,192)
  int bid = blockIdx.x;
  int chunk = bid & 63;
  int b = (bid >> 6) & 3;
  int k = bid >> 8;
  int d = threadIdx.x;
  __shared__ float sd[CHUNK * NPROJ];
  size_t dblbase = ((size_t)(k * 4 + b) * LSEQ + (size_t)chunk * CHUNK) * NPROJ;
  if (threadIdx.x < 32) {
    tdm_load_f32_to_lds(sd, x_dbl + dblbase, CHUNK * NPROJ);
    __builtin_amdgcn_s_wait_tensorcnt((short)0);
  }

  int kd = k * 192 + d;
  float dtw[6];
  #pragma unroll
  for (int r = 0; r < 6; r++) dtw[r] = dt_w[kd * 6 + r];
  float dtb = dt_b[kd];
  float Ad[16];
  #pragma unroll
  for (int n = 0; n < 16; n++) Ad[n] = -__expf(A_logs[(size_t)kd * 16 + n]);
  float Dv = Ds[kd];

  int ch = (k * 4 + b) * 192 + d;
  float h[16];
  size_t hb = ((size_t)ch * NCHUNK + chunk) * 16;
  #pragma unroll
  for (int n = 0; n < 16; n++) h[n] = hin[hb + n];

  __syncthreads();

  size_t ubase = ((size_t)(k * 4 + b) * LSEQ + (size_t)chunk * CHUNK) * 192 + d;
  for (int t = 0; t < CHUNK; t++) {
    const float* row = sd + t * NPROJ;
    float xdt = dtb;
    #pragma unroll
    for (int r = 0; r < 6; r++) xdt += row[r] * dtw[r];
    float delta = softplusf(xdt);
    float u  = bf2f(xs[ubase + (size_t)t * 192]);
    float du = delta * u;
    float y = 0.f;
    #pragma unroll
    for (int n = 0; n < 16; n++) {
      float dA = __expf(delta * Ad[n]);
      h[n] = dA * h[n] + du * row[6 + n];
      y += h[n] * row[22 + n];
    }
    out_y[ubase + (size_t)t * 192] = y + Dv * u;
  }
}

// ---------------- 9: merge 4 directions (fixed order -> deterministic) -------
__global__ void merge_dirs(const float* __restrict__ out_y,
                           float* __restrict__ ym) {     // (B,L,192)
  int idx = blockIdx.x * 256 + threadIdx.x;
  if (idx >= BL * D_INNER) return;
  int d = idx % 192;
  int p = (idx / 192) & 4095;
  int b = idx / (192 * 4096);
  int m1 = ((p & 63) << 6) + (p >> 6);
  const size_t KB = (size_t)786432;   // 4096*192
  float y = out_y[(size_t)(0 * 4 + b) * KB + (size_t)p * 192 + d]
          + out_y[(size_t)(1 * 4 + b) * KB + (size_t)m1 * 192 + d]
          + out_y[(size_t)(2 * 4 + b) * KB + (size_t)(4095 - p) * 192 + d]
          + out_y[(size_t)(3 * 4 + b) * KB + (size_t)(4095 - m1) * 192 + d];
  ym[idx] = y;
}

// ---------------- 10: SE squeeze (spatial mean of high bands) ----------------
__global__ __launch_bounds__(256)
void pool_high(const float* __restrict__ x_dwt, float* __restrict__ pooled) {
  int b   = blockIdx.x / 288;
  int chh = blockIdx.x % 288;
  int c = chh / 3;
  int s = chh % 3 + 1;
  size_t base = (size_t)((b * CDIM + c) * 4 + s) * LSEQ;
  float sum = 0.f;
  for (int i = threadIdx.x; i < LSEQ; i += 256) sum += x_dwt[base + i];
  __shared__ float red[256];
  red[threadIdx.x] = sum;
  __syncthreads();
  for (int off = 128; off > 0; off >>= 1) {
    if (threadIdx.x < off) red[threadIdx.x] += red[threadIdx.x + off];
    __syncthreads();
  }
  if (threadIdx.x == 0) pooled[b * 288 + chh] = red[0] * (1.f / 4096.f);
}

// ---------------- 11: SE excitation MLP (288->18->288, sigmoid) --------------
__global__ void se_kernel(const float* __restrict__ pooled,
                          const float* __restrict__ w1, const float* __restrict__ b1,
                          const float* __restrict__ w2, const float* __restrict__ b2,
                          float* __restrict__ attn) {
  int b = blockIdx.x;
  int t = threadIdx.x;
  __shared__ float sp[288];
  __shared__ float sh[18];
  sp[t] = pooled[b * 288 + t];
  __syncthreads();
  if (t < 18) {
    float a = b1[t];
    for (int j = 0; j < 288; j++) a += w1[t * 288 + j] * sp[j];
    sh[t] = fmaxf(a, 0.f);
  }
  __syncthreads();
  float a = b2[t];
  #pragma unroll
  for (int j = 0; j < 18; j++) a += w2[t * 18 + j] * sh[j];
  attn[b * 288 + t] = 1.f / (1.f + __expf(-a));
}

// ---------------- 12: out-layernorm + SiLU(z) gate -> bf16 -------------------
__global__ void gate_ln(const float* __restrict__ ym,
                        const float* __restrict__ xz,     // z = cols 192..383
                        const float* __restrict__ onw, const float* __restrict__ onb,
                        unsigned short* __restrict__ yg) {
  int tid = blockIdx.x * 256 + threadIdx.x;
  if (tid >= BL) return;
  const float* yb = ym + (size_t)tid * 192;
  float s = 0.f, s2 = 0.f;
  for (int d = 0; d < 192; d++) { float v = yb[d]; s += v; s2 += v * v; }
  float m   = s * (1.f / 192.f);
  float var = s2 * (1.f / 192.f) - m * m;
  float inv = rsqrtf(var + 1e-5f);
  const float* zb = xz + (size_t)tid * 384 + 192;
  for (int d = 0; d < 192; d++) {
    float yn = (yb[d] - m) * inv * onw[d] + onb[d];
    yg[(size_t)tid * 192 + d] = f2bf(yn * siluf(zb[d]));
  }
}

// ---------------- 13: inverse DWT (transposed conv, dil=2, pad=3) ------------
__global__ void idwt_kernel(const float* __restrict__ yll,   // (B,L,96)
                            const float* __restrict__ x_dwt,
                            const float* __restrict__ attn,
                            const float* __restrict__ rec_w, // (96,4,6,6)
                            float* __restrict__ out) {
  int idx = blockIdx.x * 256 + threadIdx.x;
  if (idx >= B_SZ * CDIM * 128 * 128) return;
  int xo = idx & 127;
  int yo = (idx >> 7) & 127;
  int c  = (idx >> 14) % CDIM;
  int b  = idx / (CDIM << 14);
  float asc[4];
  asc[0] = 1.f;
  asc[1] = attn[b * 288 + c * 3 + 0];
  asc[2] = attn[b * 288 + c * 3 + 1];
  asc[3] = attn[b * 288 + c * 3 + 2];
  float acc = 0.f;
  #pragma unroll
  for (int s = 0; s < 4; s++) {
    const float* w = rec_w + (size_t)(c * 4 + s) * 36;
    float scale = asc[s];
    #pragma unroll
    for (int i = 0; i < 6; i++) {
      int ti = yo + i - 3;
      if (ti < 0 || (ti & 1)) continue;
      int iy = ti >> 1;
      if (iy >= 64) continue;
      #pragma unroll
      for (int j = 0; j < 6; j++) {
        int tj = xo + j - 3;
        if (tj < 0 || (tj & 1)) continue;
        int ix = tj >> 1;
        if (ix >= 64) continue;
        float v = (s == 0)
            ? yll[((size_t)(b << 12) + (iy << 6) + ix) * 96 + c]
            : x_dwt[(size_t)((b * CDIM + c) * 4 + s) * LSEQ + (iy << 6) + ix];
        acc += w[i * 6 + j] * scale * v;
      }
    }
  }
  out[idx] = acc;
}

// ---------------------------------------------------------------------------
extern "C" void kernel_launch(void* const* d_in, const int* in_sizes, int n_in,
                              void* d_out, int out_size, void* d_ws, size_t ws_size,
                              hipStream_t stream) {
  (void)in_sizes; (void)n_in; (void)out_size; (void)ws_size;
  const float* x          = (const float*)d_in[0];
  const float* dec_f      = (const float*)d_in[1];
  const float* rec_w      = (const float*)d_in[2];
  const float* ln_w       = (const float*)d_in[3];
  const float* ln_b       = (const float*)d_in[4];
  const float* se_w1      = (const float*)d_in[5];
  const float* se_b1      = (const float*)d_in[6];
  const float* se_w2      = (const float*)d_in[7];
  const float* se_b2      = (const float*)d_in[8];
  const float* in_proj_w  = (const float*)d_in[9];
  const float* conv_w     = (const float*)d_in[10];
  const float* conv_b     = (const float*)d_in[11];
  const float* x_proj_w   = (const float*)d_in[12];
  const float* dt_w       = (const float*)d_in[13];
  const float* dt_b       = (const float*)d_in[14];
  const float* A_logs     = (const float*)d_in[15];
  const float* Ds         = (const float*)d_in[16];
  const float* onw        = (const float*)d_in[17];
  const float* onb        = (const float*)d_in[18];
  const float* out_proj_w = (const float*)d_in[19];
  float* out = (float*)d_out;

  char* ws = (char*)d_ws;
  float*          x_dwt  = (float*)(ws + 0);                     // 25.2 MB
  unsigned short* A_ll   = (unsigned short*)(ws + 25165824);     //  3.1 MB
  unsigned short* wip    = (unsigned short*)(ws + 28311552);
  unsigned short* wxp    = (unsigned short*)(ws + 28385280);
  unsigned short* wop    = (unsigned short*)(ws + 28459008);
  float*          xz     = (float*)(ws + 28495872);              // 25.2 MB
  float*          xc_act = (float*)(ws + 53661696);              // 12.6 MB
  unsigned short* xs     = (unsigned short*)(ws + 66244608);     // 25.2 MB
  float*          x_dbl  = (float*)(ws + 91410432);              // 12.6 MB
  float*          chAH   = (float*)(ws + 103993344);             // 25.2 MB
  float*          hin    = (float*)(ws + 129159168);             // 12.6 MB
  float*          ym     = (float*)(ws + 141742080);             // 12.6 MB
  unsigned short* yg     = (unsigned short*)(ws + 154324992);    //  6.3 MB
  float*          yll    = (float*)(ws + 160616448);             //  6.3 MB
  float*          pooled = (float*)(ws + 166907904);
  float*          attn   = (float*)(ws + 166912512);
  float*          out_y  = (float*)(ws + 166917120);             // 50.3 MB

  // 0. weights -> packed bf16 fragments
  prep_weights<<<360, 256, 0, stream>>>(in_proj_w, x_proj_w, out_proj_w,
                                        wip, wxp, wop);
  // 1. DWT
  dwt_kernel<<<24576, 256, 0, stream>>>(x, dec_f, x_dwt);
  // 2. LL layernorm -> bf16
  ll_layernorm<<<64, 256, 0, stream>>>(x_dwt, ln_w, ln_b, A_ll);
  // 3. in_proj: (16384,96) x (96,384); 256 M-supertiles x 24 N-tiles
  gemm_bf16_wmma<<<1536, 128, 0, stream>>>(A_ll, wip, xz,
      256, 24, 96, 96, 384, 1, 0, 0, 0);
  // 4. depthwise conv + SiLU
  conv_silu<<<16384, 192, 0, stream>>>(xz, conv_w, conv_b, xc_act);
  // 5. gather scan directions
  gather_xs<<<49152, 256, 0, stream>>>(xc_act, xs);
  // 6. x_proj (batched over 4 dirs): (16384,192) x (192,48)
  gemm_bf16_wmma<<<768, 128, 0, stream>>>(xs, wxp, x_dbl,
      256, 3, 192, 192, 48, 4,
      (long long)BL * 192, (long long)(48 * 6 * 32), (long long)BL * 48);
  // 7-9. chunked selective scan (TDM-staged x_dbl chunks)
  scan_phase1<<<1024, 192, 0, stream>>>(x_dbl, xs, dt_w, dt_b, A_logs, chAH);
  scan_phase2<<<12, 256, 0, stream>>>(chAH, hin);
  scan_phase3<<<1024, 192, 0, stream>>>(x_dbl, xs, dt_w, dt_b, A_logs, Ds,
                                        hin, out_y);
  // 10. merge directions
  merge_dirs<<<12288, 256, 0, stream>>>(out_y, ym);
  // 11-12. SE path
  pool_high<<<1152, 256, 0, stream>>>(x_dwt, pooled);
  se_kernel<<<4, 288, 0, stream>>>(pooled, se_w1, se_b1, se_w2, se_b2, attn);
  // 13. out-layernorm + gate -> bf16
  gate_ln<<<64, 256, 0, stream>>>(ym, xz, onw, onb, yg);
  // 14. out_proj: (16384,192) x (192,96)
  gemm_bf16_wmma<<<384, 128, 0, stream>>>(yg, wop, yll,
      256, 6, 192, 192, 96, 1, 0, 0, 0);
  // 15. IDWT reconstruction
  idwt_kernel<<<24576, 256, 0, stream>>>(yll, x_dwt, attn, rec_w, out);
}
